// DotProductAttention_47528108097846
// MI455X (gfx1250) — matrix operand order repaired and checked by reading, not compiled, not run on hardware
//
#include <hip/hip_runtime.h>

typedef __attribute__((ext_vector_type(16))) __bf16 v16bf;
typedef __attribute__((ext_vector_type(4)))  __bf16 v4bf;
typedef __attribute__((ext_vector_type(8)))  float  v8f;

constexpr int kB = 2;
constexpr int kH = 8;
constexpr int kS = 4096;
constexpr int kD = 64;

constexpr int QTILE  = 16;              // Q rows per wave
constexpr int WAVES  = 8;               // waves per block
constexpr int QBLOCK = QTILE * WAVES;   // 128 Q rows per block
constexpr int KCHUNK = 64;              // keys per iteration (2x WMMA K)
constexpr int NCHUNK = kS / KCHUNK;     // 64 iterations

// ds_swizzle XOR butterfly within 32 lanes: imm = (xor_mask<<10) | and_mask(0x1f)
template<int IMM>
__device__ __forceinline__ float swz_xor(float x) {
    return __builtin_bit_cast(float,
        __builtin_amdgcn_ds_swizzle(__builtin_bit_cast(int, x), IMM));
}

__global__ __launch_bounds__(256)
void DotProductAttention_fa_kernel(const float* __restrict__ Q,
                                   const float* __restrict__ K,
                                   const float* __restrict__ V,
                                   float* __restrict__ O)
{
    __shared__ __bf16 kTile[2][KCHUNK][kD];         // 16 KB (double-buffered)
    __shared__ __bf16 vTile[2][KCHUNK][kD];         // 16 KB
    __shared__ __bf16 pBuf[WAVES][QTILE][KCHUNK];   // 16 KB (per-wave P bounce)

    const int tid  = threadIdx.x;
    const int wave = tid >> 5;
    const int lane = tid & 31;
    const int hf   = lane >> 4;   // 16-lane half
    const int l16  = lane & 15;

    const int bh    = blockIdx.y;                        // b*H + h
    const int qBase = blockIdx.x * QBLOCK + wave * QTILE;

    const float* qp = Q + (size_t)bh * kS * kD;
    const float* kp = K + (size_t)bh * kS * kD;
    const float* vp = V + (size_t)bh * kS * kD;
    float*       op = O + (size_t)bh * kS * kD;

    // ---- Q as two bf16 A-fragments, PRE-SCALED by 1/16 (reference's score/16).
    // A 16x32 bf16 operand, contiguous k order chosen here: lane holds row M=l16; elem e -> Kdim = e + 16*hf
    v16bf qa0, qa1;
    {
        const float* qrow = qp + (size_t)(qBase + l16) * kD;
        #pragma unroll
        for (int e = 0; e < 16; ++e) {
            int kk = e + 16 * hf;
            qa0[e] = (__bf16)(qrow[kk]      * 0.0625f);
            qa1[e] = (__bf16)(qrow[32 + kk] * 0.0625f);
        }
    }

    // all-ones B-fragment: folds the softmax row-sum into the matrix pipe
    v16bf ones;
    #pragma unroll
    for (int e = 0; e < 16; ++e) ones[e] = (__bf16)1.0f;

    // Accumulators (C layout: row = v + 8*hf). oacc = O tiles, lacc = row sums.
    v8f oacc[4], lacc = (v8f){};
    #pragma unroll
    for (int t = 0; t < 4; ++t) oacc[t] = (v8f){};

    float mrow[8];
    #pragma unroll
    for (int v = 0; v < 8; ++v) mrow[v] = -INFINITY;

    // ---- Software-pipelined K/V staging (prefetch regs -> cvt -> LDS) ----
    float4 kReg[4], vReg[4];
    auto stageLoad = [&](int kBase) {
        #pragma unroll
        for (int i = 0; i < 4; ++i) {
            int e = (tid + i * 256) * 4, krow = e >> 6, d = e & 63;
            kReg[i] = *(const float4*)&kp[(size_t)(kBase + krow) * kD + d];
            vReg[i] = *(const float4*)&vp[(size_t)(kBase + krow) * kD + d];
        }
    };
    auto stageStore = [&](int buf) {
        #pragma unroll
        for (int i = 0; i < 4; ++i) {
            int e = (tid + i * 256) * 4, krow = e >> 6, d = e & 63;
            v4bf kb; kb[0]=(__bf16)kReg[i].x; kb[1]=(__bf16)kReg[i].y;
                     kb[2]=(__bf16)kReg[i].z; kb[3]=(__bf16)kReg[i].w;
            v4bf vb; vb[0]=(__bf16)vReg[i].x; vb[1]=(__bf16)vReg[i].y;
                     vb[2]=(__bf16)vReg[i].z; vb[3]=(__bf16)vReg[i].w;
            *(v4bf*)&kTile[buf][krow][d] = kb;
            *(v4bf*)&vTile[buf][krow][d] = vb;
        }
    };

    stageLoad(0);
    stageStore(0);
    __syncthreads();

    for (int c = 0; c < NCHUNK; ++c) {
        const int cur = c & 1;
        const int nxt = cur ^ 1;

        // issue next chunk's global loads NOW; compute hides their latency
        if (c + 1 < NCHUNK) stageLoad((c + 1) * KCHUNK);

        // ---- S = (Q/16) K^T : 4 key-subtiles x 2 D-steps = 8 WMMA ----
        // B 32x16 operand, same contiguous k order as A: lane holds col N=l16; elem e -> Kdim = e + 16*hf
        v8f s[4];
        #pragma unroll
        for (int sub = 0; sub < 4; ++sub) {
            v16bf kbA, kbB;
            #pragma unroll
            for (int e = 0; e < 16; ++e) {
                int d = e + 16 * hf;
                kbA[e] = kTile[cur][sub * 16 + l16][d];
                kbB[e] = kTile[cur][sub * 16 + l16][32 + d];
            }
            v8f acc = (v8f){};
            acc = __builtin_amdgcn_wmma_f32_16x16x32_bf16(false, qa0, false, kbA, (short)0, acc, false, false);
            acc = __builtin_amdgcn_wmma_f32_16x16x32_bf16(false, qa1, false, kbB, (short)0, acc, false, false);
            s[sub] = acc;
        }

        // ---- Online softmax: per-row max via ds_swizzle XOR butterfly ----
        #pragma unroll
        for (int v = 0; v < 8; ++v) {
            float a0 = s[0][v], a1 = s[1][v], a2 = s[2][v], a3 = s[3][v];
            float rm = fmaxf(fmaxf(a0, a1), fmaxf(a2, a3));
            rm = fmaxf(rm, swz_xor<0x041f>(rm));   // xor 1
            rm = fmaxf(rm, swz_xor<0x081f>(rm));   // xor 2
            rm = fmaxf(rm, swz_xor<0x101f>(rm));   // xor 4
            rm = fmaxf(rm, swz_xor<0x201f>(rm));   // xor 8 (stays in 16-lane half)
            float mn = fmaxf(mrow[v], rm);
            float sf = __expf(mrow[v] - mn);
            mrow[v] = mn;
            lacc[v] *= sf;
            #pragma unroll
            for (int t = 0; t < 4; ++t) oacc[t][v] *= sf;

            int row = v + 8 * hf;
            pBuf[wave][row][l16]      = (__bf16)__expf(a0 - mn);
            pBuf[wave][row][16 + l16] = (__bf16)__expf(a1 - mn);
            pBuf[wave][row][32 + l16] = (__bf16)__expf(a2 - mn);
            pBuf[wave][row][48 + l16] = (__bf16)__expf(a3 - mn);
        }

        // ---- P back as two 16x32 A-fragments (same-wave LDS, in-order) ----
        v16bf pa0, pa1;
        #pragma unroll
        for (int e = 0; e < 16; ++e) {
            int kk = e + 16 * hf;
            pa0[e] = pBuf[wave][l16][kk];
            pa1[e] = pBuf[wave][l16][32 + kk];
        }

        // ---- O += P V : 4 d-tiles x 2 key-steps; l += P * ones ----
        #pragma unroll
        for (int t = 0; t < 4; ++t) {
            v16bf vb0, vb1;
            #pragma unroll
            for (int e = 0; e < 16; ++e) {
                int kr = e + 16 * hf;
                vb0[e] = vTile[cur][kr][t * 16 + l16];
                vb1[e] = vTile[cur][32 + kr][t * 16 + l16];
            }
            oacc[t] = __builtin_amdgcn_wmma_f32_16x16x32_bf16(false, pa0, false, vb0, (short)0, oacc[t], false, false);
            oacc[t] = __builtin_amdgcn_wmma_f32_16x16x32_bf16(false, pa1, false, vb1, (short)0, oacc[t], false, false);
        }
        lacc = __builtin_amdgcn_wmma_f32_16x16x32_bf16(false, pa0, false, ones, (short)0, lacc, false, false);
        lacc = __builtin_amdgcn_wmma_f32_16x16x32_bf16(false, pa1, false, ones, (short)0, lacc, false, false);

        // convert + store next chunk behind the WMMAs, then fence buffers
        if (c + 1 < NCHUNK) stageStore(nxt);
        __syncthreads();
    }

    // ---- Epilogue: O / l; [B,H,S,D] flat == reshape(S,B,H,D) flat ----
    #pragma unroll
    for (int v = 0; v < 8; ++v) {
        int row = qBase + v + 8 * hf;
        float invl = 1.0f / lacc[v];
        #pragma unroll
        for (int t = 0; t < 4; ++t) {
            op[(size_t)row * kD + t * 16 + l16] = oacc[t][v] * invl;
        }
    }
}

extern "C" void kernel_launch(void* const* d_in, const int* in_sizes, int n_in,
                              void* d_out, int out_size, void* d_ws, size_t ws_size,
                              hipStream_t stream) {
    const float* Q = (const float*)d_in[0];
    const float* K = (const float*)d_in[1];
    const float* V = (const float*)d_in[2];
    // d_in[3] (adj) unused by the reference module
    float* O = (float*)d_out;

    dim3 grid(kS / QBLOCK, kB * kH);   // (32, 16)
    dim3 block(256);
    DotProductAttention_fa_kernel<<<grid, block, 0, stream>>>(Q, K, V, O);
}
